// SGA_71038759075971
// MI455X (gfx1250) — matrix operand order repaired and checked by reading, not compiled, not run on hardware
//
#include <hip/hip_runtime.h>

// ---------------------------------------------------------------------------
// XCA-style channel attention for MI455X (gfx1250, wave32, WMMA bf16).
//   t   = w_qv @ x            (bf16 WMMA GEMM, f32 out, async-LDS staging)
//   v   = dwconv3x3(t)        (VALU, bf16 out)
//   G   = q q^T ; P = softmax(G/sqrt(384))   (bf16 WMMA + LDS f32 atomics,
//                                             wave-private async-LDS staging)
//   out = w_proj @ (P @ v)    (fused bf16 WMMA GEMM pair)
// Workspace layout (bytes): t_f32[0, 201326592) xbf[.., +100663296)
//   vbf(+100663296) wqvb(+294912) wprb(+294912) Pbf(+393216)  total ~385 MB.
// ---------------------------------------------------------------------------

#define DIMC 384
#define HEADS 8
#define CH 48            // channels per head
#define HW 128
#define SPAT (HW * HW)   // 16384
#define NB 8

typedef __attribute__((ext_vector_type(16))) __bf16 v16bf;
typedef __attribute__((ext_vector_type(8)))  float  v8f;

union Frag {
  v16bf          v;
  unsigned short h[16];
  unsigned int   u[8];
  uint4          q[2];
};

__device__ __forceinline__ unsigned short f2bf(float x) {
  union { float f; unsigned int u; } c; c.f = x;
  unsigned int r = c.u + 0x7fffu + ((c.u >> 16) & 1u);   // round-nearest-even
  return (unsigned short)(r >> 16);
}

__device__ __forceinline__ v8f wmma_bf16(const Frag& a, const Frag& b, v8f c) {
  return __builtin_amdgcn_wmma_f32_16x16x32_bf16(false, a.v, false, b.v,
                                                 (short)0, c, false, false);
}

// --- CDNA5 async global->LDS copy (ASYNCcnt tracked), per-lane 16B --------
__device__ __forceinline__ void async_ld_b128(unsigned lds_off, const void* g) {
  asm volatile("global_load_async_to_lds_b128 %0, %1, off"
               :: "v"(lds_off), "v"(g) : "memory");
}
__device__ __forceinline__ void wait_async0() {
  asm volatile("s_wait_asynccnt 0x0" ::: "memory");
}
__device__ __forceinline__ void wait_ds0() {
  asm volatile("s_wait_dscnt 0x0" ::: "memory");
}
__device__ __forceinline__ unsigned lds_off_of(const void* p) {
  return (unsigned)(size_t)p;          // generic addr low 32 bits = LDS offset
}

// -------------------------- fp32 -> bf16 convert ---------------------------
__global__ void __launch_bounds__(256) k_cvt4(const float* __restrict__ s,
                                              unsigned short* __restrict__ d,
                                              int n4) {
  int i = blockIdx.x * 256 + threadIdx.x;
  if (i >= n4) return;
  float4 f = reinterpret_cast<const float4*>(s)[i];
  unsigned long long p = (unsigned long long)f2bf(f.x)
                       | ((unsigned long long)f2bf(f.y) << 16)
                       | ((unsigned long long)f2bf(f.z) << 32)
                       | ((unsigned long long)f2bf(f.w) << 48);
  reinterpret_cast<unsigned long long*>(d)[i] = p;
}

// ------------------- GEMM: t[b,o,s] = sum_c w_qv[o,c] x[b,c,s] -------------
// grid = (SPAT/64, NB), 256 threads = 8 waves; wave w owns o-rows [48w,48w+48)
__global__ void __launch_bounds__(256) k_gemm_qv(
    const unsigned short* __restrict__ xb,   // [B][384][S] bf16
    const unsigned short* __restrict__ wb,   // [384][384] bf16
    float* __restrict__ t)                   // [B][384][S] f32
{
  const int b     = blockIdx.y;
  const int s0    = blockIdx.x * 64;
  const int wave  = threadIdx.x >> 5;
  const int lane  = threadIdx.x & 31;
  const int oBase = wave * CH;

  __shared__ __align__(16) unsigned int ldsX[32 * 33]; // 32 c-rows x 64 s halves

  v8f acc[3][4] = {};

  const int srow = threadIdx.x >> 3;   // 0..31  (c row within tile)
  const int schk = threadIdx.x & 7;    // 0..7   (8 halves each)
  const unsigned stg_lds = lds_off_of(ldsX + srow * 33 + schk * 4);

  for (int kk = 0; kk < DIMC / 32; ++kk) {
    const int c0 = kk * 32;
    // async-stage X[c0..c0+31][s0..s0+64) as bf16 straight into LDS
    async_ld_b128(stg_lds,
                  xb + (size_t)(b * DIMC + c0 + srow) * SPAT + s0 + schk * 8);
    wait_async0();
    __syncthreads();

    Frag A[3];
#pragma unroll
    for (int i = 0; i < 3; ++i) {
      const unsigned short* p =
          wb + (size_t)(oBase + 16 * i + (lane & 15)) * DIMC + c0 + 8 * (lane >> 4);
      A[i].q[0] = *reinterpret_cast<const uint4*>(p);
      A[i].q[1] = *reinterpret_cast<const uint4*>(p + 16);
    }
#pragma unroll
    for (int n0 = 0; n0 < 4; ++n0) {
      Frag B;
#pragma unroll
      for (int j = 0; j < 16; ++j) B.h[j] = reinterpret_cast<const unsigned short*>(ldsX)[(8 * (lane >> 4) + (j & 7) + 16 * (j >> 3)) * 66 + n0 * 16 + (lane & 15)];
#pragma unroll
      for (int i = 0; i < 3; ++i) acc[i][n0] = wmma_bf16(A[i], B, acc[i][n0]);
    }
    __syncthreads();   // compiler fronts this with s_wait_dscnt: WAR-safe
  }

#pragma unroll
  for (int i = 0; i < 3; ++i)
#pragma unroll
    for (int n0 = 0; n0 < 4; ++n0)
#pragma unroll
      for (int g = 0; g < 8; ++g) {
        const int o = oBase + 16 * i + g + 8 * (lane >> 4);
        const int s = s0 + n0 * 16 + (lane & 15);
        t[((size_t)b * DIMC + o) * SPAT + s] = acc[i][n0][g];
      }
}

// ---------------------- depthwise 3x3 conv, SAME padding -------------------
__global__ void __launch_bounds__(256) k_dwconv(
    const float* __restrict__ t, const float* __restrict__ wdw,
    unsigned short* __restrict__ vb)
{
  const size_t i = (size_t)blockIdx.x * 256 + threadIdx.x;
  const int xw = (int)(i & (HW - 1));
  const int y  = (int)((i >> 7) & (HW - 1));
  const int cb = (int)(i >> 14);        // b*384 + c
  const int c  = cb % DIMC;
  const float* p = t + (size_t)cb * SPAT;
  const float* w = wdw + c * 9;
  float acc = 0.0f;
#pragma unroll
  for (int dy = -1; dy <= 1; ++dy) {
    const int yy = y + dy;
    if (yy < 0 || yy >= HW) continue;
#pragma unroll
    for (int dx = -1; dx <= 1; ++dx) {
      const int xx = xw + dx;
      if (xx < 0 || xx >= HW) continue;
      acc += p[yy * HW + xx] * w[(dy + 1) * 3 + (dx + 1)];
    }
  }
  vb[i] = f2bf(acc);
}

// ------------- Gram (q q^T) + softmax -> P bf16, padded [64][48][64] -------
// grid = 64 (one block per (b,head)); wave w reduces s-range [2048w, 2048w+2048)
__global__ void __launch_bounds__(256) k_gram_softmax(
    const unsigned short* __restrict__ xb,   // [B][384][S] bf16
    unsigned short* __restrict__ Pb)         // [64][48][64] bf16
{
  const int bh   = blockIdx.x;
  const int wave = threadIdx.x >> 5;
  const int lane = threadIdx.x & 31;

  __shared__ float G[CH * CH];
  __shared__ __align__(16) unsigned short tile[8][CH * 40];  // 48x32 bf16/wave

  for (int idx = threadIdx.x; idx < CH * CH; idx += 256) G[idx] = 0.0f;
  __syncthreads();

  const unsigned short* xh = xb + (size_t)bh * CH * SPAT;

  v8f acc[3][3] = {};
  unsigned short* tl = tile[wave];

  for (int step = 0; step < 64; ++step) {
    const int s0 = wave * 2048 + step * 32;
    // wave-private async staging: 48 rows x 64B -> LDS (no barrier needed).
    // DS reads and async LDS writes are mutually unordered, so drain DS first.
    wait_ds0();
#pragma unroll
    for (int tix = 0; tix < 6; ++tix) {
      const int idx = lane + 32 * tix;   // 0..191 = 48 rows x 4 chunks
      const int r   = idx >> 2;
      const int cc  = idx & 3;
      async_ld_b128(lds_off_of(tl + r * 40 + cc * 8),
                    xh + (size_t)r * SPAT + s0 + cc * 8);
    }
    wait_async0();

    Frag A[3];
#pragma unroll
    for (int i = 0; i < 3; ++i) {
      const unsigned short* p = tl + (16 * i + (lane & 15)) * 40 + 8 * (lane >> 4);
      A[i].q[0] = *reinterpret_cast<const uint4*>(p);
      A[i].q[1] = *reinterpret_cast<const uint4*>(p + 16);
    }
#pragma unroll
    for (int j = 0; j < 3; ++j) {
      Frag B;   // B[k][n] = q[16j+n][s0+k]  (lane n = lane&15, table-order k)
      const unsigned short* pb = tl + (16 * j + (lane & 15)) * 40 + 8 * (lane >> 4);
      B.q[0] = *reinterpret_cast<const uint4*>(pb);
      B.q[1] = *reinterpret_cast<const uint4*>(pb + 16);
#pragma unroll
      for (int i = 0; i < 3; ++i) acc[i][j] = wmma_bf16(A[i], B, acc[i][j]);
    }
  }

  // cross-wave reduction of the 48x48 Gram via LDS float atomics
#pragma unroll
  for (int i = 0; i < 3; ++i)
#pragma unroll
    for (int j = 0; j < 3; ++j)
#pragma unroll
      for (int g = 0; g < 8; ++g) {
        const int r = 16 * i + g + 8 * (lane >> 4);
        const int d = 16 * j + (lane & 15);
        atomicAdd(&G[r * CH + d], acc[i][j][g]);
      }
  __syncthreads();

  if (threadIdx.x < CH) {
    const float temp = 0.05103103630798287f;  // 1/sqrt(384)
    const int r = threadIdx.x;
    float m = -3.4e38f;
    for (int d = 0; d < CH; ++d) m = fmaxf(m, G[r * CH + d] * temp);
    float sum = 0.0f;
    for (int d = 0; d < CH; ++d) {
      const float e = __expf(G[r * CH + d] * temp - m);
      G[r * CH + d] = e;
      sum += e;
    }
    const float inv = 1.0f / sum;
    unsigned short* dst = Pb + ((size_t)bh * CH + r) * 64;
    for (int d = 0; d < CH; ++d)  dst[d] = f2bf(G[r * CH + d] * inv);
    for (int d = CH; d < 64; ++d) dst[d] = 0;   // K-pad for 2x K=32 WMMA steps
  }
}

// --------------- fused: out = w_proj @ (P @ v), per 32-wide s tile ---------
// grid = (SPAT/32, NB); stage1: wave == head computes P@v -> LDS bf16 tile;
// stage2: wave w projects o-rows [48w,48w+48) straight to d_out.
__global__ void __launch_bounds__(256) k_attn_proj(
    const unsigned short* __restrict__ Pb,   // [64][48][64] bf16
    const unsigned short* __restrict__ vb,   // [B][384][S]  bf16
    const unsigned short* __restrict__ wpb,  // [384][384]   bf16
    float* __restrict__ out)                 // [B][384][S]  f32
{
  const int b    = blockIdx.y;
  const int s0   = blockIdx.x * 32;
  const int wave = threadIdx.x >> 5;
  const int lane = threadIdx.x & 31;

  __shared__ __align__(16) unsigned short A2[DIMC * 34];  // attn-out bf16

  {   // ---- stage 1: attn_out rows [48h, 48h+48) for head h = wave ----
    const int h = wave;
    const unsigned short* Ph = Pb + (size_t)(b * HEADS + h) * CH * 64;
    const unsigned short* Vh = vb + ((size_t)b * DIMC + h * CH) * SPAT;

    v8f acc[3][2] = {};
#pragma unroll
    for (int kkk = 0; kkk < 2; ++kkk) {
      const int d0 = kkk * 32;
      Frag A[3];
#pragma unroll
      for (int i = 0; i < 3; ++i) {
        const unsigned short* p =
            Ph + (size_t)(16 * i + (lane & 15)) * 64 + d0 + 8 * (lane >> 4);
        A[i].q[0] = *reinterpret_cast<const uint4*>(p);
        A[i].q[1] = *reinterpret_cast<const uint4*>(p + 16);
      }
#pragma unroll
      for (int n0 = 0; n0 < 2; ++n0) {
        Frag B;
#pragma unroll
        for (int e = 0; e < 16; ++e) {
          const int d = d0 + 8 * (lane >> 4) + (e & 7) + 16 * (e >> 3);
          B.h[e] = (d < CH) ? Vh[(size_t)d * SPAT + s0 + n0 * 16 + (lane & 15)] : (unsigned short)0;
        }
#pragma unroll
        for (int i = 0; i < 3; ++i) acc[i][n0] = wmma_bf16(A[i], B, acc[i][n0]);
      }
    }
#pragma unroll
    for (int i = 0; i < 3; ++i)
#pragma unroll
      for (int n0 = 0; n0 < 2; ++n0)
#pragma unroll
        for (int g = 0; g < 8; ++g) {
          const int row = h * CH + 16 * i + g + 8 * (lane >> 4);
          const int col = n0 * 16 + (lane & 15);
          A2[row * 34 + col] = f2bf(acc[i][n0][g]);
        }
  }
  __syncthreads();

  {   // ---- stage 2: out[o,s] = sum_c w_proj[o,c] * attn_out[c,s] ----
    const int oBase = wave * CH;
    const unsigned int* a2u = reinterpret_cast<const unsigned int*>(A2);
    v8f acc[3][2] = {};
    for (int kk = 0; kk < DIMC / 32; ++kk) {
      const int c0 = kk * 32;
      Frag A[3];
#pragma unroll
      for (int i = 0; i < 3; ++i) {
        const unsigned short* p =
            wpb + (size_t)(oBase + 16 * i + (lane & 15)) * DIMC + c0 + 8 * (lane >> 4);
        A[i].q[0] = *reinterpret_cast<const uint4*>(p);
        A[i].q[1] = *reinterpret_cast<const uint4*>(p + 16);
      }
#pragma unroll
      for (int n0 = 0; n0 < 2; ++n0) {
        Frag B;
#pragma unroll
        for (int j = 0; j < 16; ++j) B.h[j] = A2[(c0 + 8 * (lane >> 4) + (j & 7) + 16 * (j >> 3)) * 34 + n0 * 16 + (lane & 15)];
#pragma unroll
        for (int i = 0; i < 3; ++i) acc[i][n0] = wmma_bf16(A[i], B, acc[i][n0]);
      }
    }
#pragma unroll
    for (int i = 0; i < 3; ++i)
#pragma unroll
      for (int n0 = 0; n0 < 2; ++n0)
#pragma unroll
        for (int g = 0; g < 8; ++g) {
          const int o = oBase + 16 * i + g + 8 * (lane >> 4);
          const int s = s0 + n0 * 16 + (lane & 15);
          out[((size_t)b * DIMC + o) * SPAT + s] = acc[i][n0][g];
        }
  }
}

// ---------------------------------------------------------------------------
extern "C" void kernel_launch(void* const* d_in, const int* in_sizes, int n_in,
                              void* d_out, int out_size, void* d_ws, size_t ws_size,
                              hipStream_t stream) {
  (void)in_sizes; (void)n_in; (void)out_size; (void)ws_size;
  const float* x      = (const float*)d_in[0];
  const float* w_qv   = (const float*)d_in[1];
  const float* w_dw   = (const float*)d_in[2];
  const float* w_proj = (const float*)d_in[3];
  float* out = (float*)d_out;

  char* ws = (char*)d_ws;                       // needs ~385 MB scratch
  const size_t nX = (size_t)NB * DIMC * SPAT;   // 50,331,648 elements
  float*          t_f32 = (float*)(ws);
  unsigned short* xbf   = (unsigned short*)(ws + 201326592ull);
  unsigned short* vbf   = (unsigned short*)(ws + 301989888ull);
  unsigned short* wqvb  = (unsigned short*)(ws + 402653184ull);
  unsigned short* wprb  = (unsigned short*)(ws + 402948096ull);
  unsigned short* Pbf   = (unsigned short*)(ws + 403243008ull);

  k_cvt4<<<(int)(nX / 4 / 256), 256, 0, stream>>>(x, xbf, (int)(nX / 4));
  k_cvt4<<<(DIMC * DIMC / 4 + 255) / 256, 256, 0, stream>>>(w_qv,   wqvb, DIMC * DIMC / 4);
  k_cvt4<<<(DIMC * DIMC / 4 + 255) / 256, 256, 0, stream>>>(w_proj, wprb, DIMC * DIMC / 4);

  k_gemm_qv     <<<dim3(SPAT / 64, NB), 256, 0, stream>>>(xbf, wqvb, t_f32);
  k_dwconv      <<<(int)(nX / 256),     256, 0, stream>>>(t_f32, w_dw, vbf);
  k_gram_softmax<<<NB * HEADS,          256, 0, stream>>>(xbf, Pbf);
  k_attn_proj   <<<dim3(SPAT / 32, NB), 256, 0, stream>>>(Pbf, vbf, wprb, out);
}
